// DecodePredictions_1486058684745
// MI455X (gfx1250) — compile-verified, matrix operand order checked by reading, not run on hardware
//
#include <hip/hip_runtime.h>
#include <hip/hip_bf16.h>

// ---------------------------------------------------------------------------
// RetinaNet decode + per-class NMS + global top-K for MI455X (gfx1250).
//  K0 decode      : anchors recomputed analytically, boxes + transposed logits
//  K1 topk        : per (image,class) top-500 via binary-search select + bitonic
//  K2 nms         : TDM async load of candidates to LDS, WMMA-assisted IoU tiles,
//                   wave32 ballot adjacency bitmasks, 1-wave suppression scan
//  K3 final       : per-image 16384-wide bitonic sort in LDS -> top-1000 outputs
// ---------------------------------------------------------------------------

#define BATCH   8
#define NANCH   76725
#define NCLS    18
#define KSEL    500
#define KPAD    512
#define MAXDET  1000
#define FLAT_C  (NCLS * KSEL)          // 9000 candidates per image
#define SORTN   16384                  // power-of-two pad for final bitonic

typedef _Float16 v16h __attribute__((ext_vector_type(16)));
typedef float    v8f  __attribute__((ext_vector_type(8)));
typedef unsigned int u32x4 __attribute__((ext_vector_type(4)));
typedef int          i32x4 __attribute__((ext_vector_type(4)));
typedef int          i32x8 __attribute__((ext_vector_type(8)));

#define AREA_SCL     (1.0f / 65536.0f)  // keep areas inside f16 range for WMMA
#define AREA_SCL_INV 65536.0f

// order-preserving float -> uint key (ascending)
__device__ __forceinline__ unsigned int fkey(float f) {
    unsigned int u = __float_as_uint(f);
    return (u & 0x80000000u) ? ~u : (u | 0x80000000u);
}
__device__ __forceinline__ float fkey_inv(unsigned int k) {
    unsigned int u = (k & 0x80000000u) ? (k & 0x7FFFFFFFu) : ~k;
    return __uint_as_float(u);
}

// ---------------------------------------------------------------------------
// Tensor Data Mover: 1-D contiguous global -> LDS DMA (CDNA5 TDM path).
// D# built per CDNA5 ISA. data_size=4B, 1 row tile.
// ---------------------------------------------------------------------------
__device__ __forceinline__ void tdm_load_1d(unsigned int lds_addr,
                                            const void* gptr,
                                            unsigned int nelem /* dwords, <=65535 */) {
    unsigned long long ga = (unsigned long long)(size_t)gptr;
    u32x4 g0;
    g0[0] = 1u;                                        // count=1, user mode
    g0[1] = lds_addr;                                  // LDS byte address
    g0[2] = (unsigned int)ga;                          // global_addr[31:0]
    g0[3] = (unsigned int)((ga >> 32) & 0x01FFFFFFull) // global_addr[56:32]
          | (2u << 30);                                // type=2 ("image")
    i32x8 g1;
    g1[0] = (int)(2u << 16);                           // data_size=4B
    g1[1] = (int)((nelem & 0xFFFFu) << 16);            // tensor_dim0[15:0]
    g1[2] = (int)((nelem >> 16) | (1u << 16));         // tensor_dim0[31:16], tensor_dim1=1
    g1[3] = (int)((nelem & 0xFFFFu) << 16);            // tile_dim0 = nelem
    g1[4] = 0;                                         // tile_dim1/2 unused
    g1[5] = (int)nelem;                                // tensor_dim0_stride[31:0]
    g1[6] = 0;
    g1[7] = 0;
    i32x4 gz; gz[0] = gz[1] = gz[2] = gz[3] = 0;
#if __has_include(<hip/amd_detail/amd_gfx1250_TDM.h>)
    i32x8 gz8; gz8[0]=gz8[1]=gz8[2]=gz8[3]=gz8[4]=gz8[5]=gz8[6]=gz8[7]=0;
    __builtin_amdgcn_tensor_load_to_lds(g0, g1, gz, gz, gz8, 0);
#else
    __builtin_amdgcn_tensor_load_to_lds(g0, g1, gz, gz, 0);
#endif
}

// ---------------------------------------------------------------------------
// K0: decode anchors + boxes, transpose class logits to [B][C][N]
// ---------------------------------------------------------------------------
__global__ __launch_bounds__(256)
void decode_kernel(const float* __restrict__ pred,
                   float* __restrict__ boxes,
                   float* __restrict__ scoresT) {
    int gid = blockIdx.x * blockDim.x + threadIdx.x;
    if (gid >= BATCH * NANCH) return;
    int b = gid / NANCH, i = gid % NANCH;

    int base, fw; float stride, area;
    if      (i < 57600) { base = 0;     fw = 80; stride =   8.f; area =   1024.f; }
    else if (i < 72000) { base = 57600; fw = 40; stride =  16.f; area =   4096.f; }
    else if (i < 75600) { base = 72000; fw = 20; stride =  32.f; area =  16384.f; }
    else if (i < 76500) { base = 75600; fw = 10; stride =  64.f; area =  65536.f; }
    else                { base = 76500; fw =  5; stride = 128.f; area = 262144.f; }
    int li = i - base;
    int k = li % 9, cell = li / 9;
    float acx = ((float)(cell % fw) + 0.5f) * stride;
    float acy = ((float)(cell / fw) + 0.5f) * stride;
    int ri = k / 3, si = k % 3;
    float r = (ri == 0) ? 0.5f : (ri == 1 ? 1.0f : 2.0f);
    float s = (si == 0) ? 1.0f : (si == 1 ? 1.2599210498948732f : 1.5874010519681994f);
    float aw = s * sqrtf(area * r);
    float ah = s * sqrtf(area / r);

    const float* p = pred + (size_t)gid * 22;
    __builtin_prefetch(p + 22 * 256, 0, 1);   // global_prefetch_b8 for next tile

    float t0 = p[0] * 0.1f, t1 = p[1] * 0.1f;
    float t2 = p[2] * 0.2f, t3 = p[3] * 0.2f;
    float cx = t0 * aw + acx, cy = t1 * ah + acy;
    float w  = expf(t2) * aw, h  = expf(t3) * ah;
    float4 bx = make_float4(cx - 0.5f * w, cy - 0.5f * h, cx + 0.5f * w, cy + 0.5f * h);
    ((float4*)boxes)[gid] = bx;

    #pragma unroll
    for (int c = 0; c < NCLS; ++c)
        scoresT[((size_t)b * NCLS + c) * NANCH + i] = p[4 + c];
}

// ---------------------------------------------------------------------------
// K1: top-500 per (image,class). Binary search the 500th-largest key, gather,
// bitonic sort 512 in LDS by (key desc, index asc), emit sorted candidates.
// ---------------------------------------------------------------------------
__global__ __launch_bounds__(1024)
void topk_kernel(const float* __restrict__ scoresT,
                 const float* __restrict__ boxes,
                 float* __restrict__ cand_box,
                 float* __restrict__ cand_score) {
    int bc = blockIdx.x;
    int b  = bc / NCLS;
    const float* sc = scoresT + (size_t)bc * NANCH;
    int t = threadIdx.x, lane = t & 31;

    __shared__ int red;
    __shared__ int scnt;
    __shared__ unsigned int skey[KPAD];
    __shared__ unsigned int sidx[KPAD];

    // --- bitwise binary search: T = max x with count(key >= x) >= KSEL ---
    unsigned int T = 0;
    for (int bit = 31; bit >= 0; --bit) {
        unsigned int test = T | (1u << bit);
        if (t == 0) red = 0;
        __syncthreads();
        int local = 0;
        for (int i = t; i < NANCH; i += 1024)
            local += (fkey(sc[i]) >= test) ? 1 : 0;
        for (int off = 16; off > 0; off >>= 1)
            local += __shfl_down(local, off, 32);
        if (lane == 0) atomicAdd(&red, local);
        __syncthreads();
        if (red >= KSEL) T = test;
        __syncthreads();
    }

    // --- gather all key >= T (>=500 of them; ties beyond 512 dropped) ---
    if (t == 0) scnt = 0;
    __syncthreads();
    for (int i = t; i < NANCH; i += 1024) {
        unsigned int kk = fkey(sc[i]);
        if (kk >= T) {
            int p = atomicAdd(&scnt, 1);
            if (p < KPAD) { skey[p] = kk; sidx[p] = (unsigned int)i; }
        }
    }
    __syncthreads();
    if (t >= scnt && t < KPAD) { skey[t] = 0; sidx[t] = 0xFFFFFFFFu; }
    __syncthreads();

    // --- bitonic sort 512: key descending, index ascending on ties ---
    for (unsigned int k = 2; k <= KPAD; k <<= 1) {
        for (unsigned int j = k >> 1; j > 0; j >>= 1) {
            for (unsigned int idx = t; idx < KPAD; idx += 1024) {
                unsigned int ixj = idx ^ j;
                if (ixj > idx) {
                    bool desc = ((idx & k) == 0);
                    unsigned int ka = skey[idx], kb = skey[ixj];
                    unsigned int ia = sidx[idx], ib = sidx[ixj];
                    bool a_first = (ka > kb) || (ka == kb && ia < ib);
                    if (desc ? !a_first : a_first) {
                        skey[idx] = kb; skey[ixj] = ka;
                        sidx[idx] = ib; sidx[ixj] = ia;
                    }
                }
            }
            __syncthreads();
        }
    }

    if (t < KSEL) {
        unsigned int kk = skey[t], i = sidx[t];
        bool ok = (i != 0xFFFFFFFFu);
        float logit = ok ? fkey_inv(kk) : -30.f;
        float prob  = 1.f / (1.f + expf(-logit));
        cand_score[(size_t)bc * KSEL + t] = ok ? prob : 0.f;
        float4 bx = ok ? ((const float4*)boxes)[(size_t)b * NANCH + i]
                       : make_float4(0.f, 0.f, 0.f, 0.f);
        ((float4*)cand_box)[(size_t)bc * KSEL + t] = bx;
    }
}

// ---------------------------------------------------------------------------
// K2: NMS per (image,class). TDM async load candidates -> LDS; adjacency via
// 16x16 WMMA tiles (union outer-sum on matrix pipe) + VALU intersections +
// wave32 ballot; greedy suppression scan on wave 0 with bitmask keep words.
// Threshold test is division-free: inter/max(uni,eps) > 0.5
//                              <=>  inter > 0.5*max(uni,eps).
// ---------------------------------------------------------------------------
__global__ __launch_bounds__(512)
void nms_kernel(const float* __restrict__ cand_box,
                float* __restrict__ cand_score) {
    int bc = blockIdx.x;
    int t = threadIdx.x, lane = t & 31, wv = t >> 5;

    __shared__ float4 sbox[KPAD];
    __shared__ float  sval[KPAD];
    __shared__ float  sarea[KPAD];
    __shared__ unsigned int adj[KPAD][16];
    __shared__ unsigned int skept[KPAD];

    if (t < 32) {  // wave 0 issues the async tensor DMAs
        tdm_load_1d((unsigned int)(size_t)(void*)&sbox[0],
                    cand_box + (size_t)bc * KSEL * 4, KSEL * 4);
        tdm_load_1d((unsigned int)(size_t)(void*)&sval[0],
                    cand_score + (size_t)bc * KSEL, KSEL);
        __builtin_amdgcn_s_wait_tensorcnt(0);
    }
    if (t >= KSEL) { sbox[t] = make_float4(0.f, 0.f, 0.f, 0.f); sval[t] = -1.f; }
    __syncthreads();

    float4 bt = sbox[t];
    sarea[t] = (bt.z - bt.x) * (bt.w - bt.y);
    #pragma unroll
    for (int q = 0; q < 16; ++q) adj[t][q] = 0;
    __syncthreads();

    // 32x32 grid of 16x16 tiles over the 512x512 (padded) pair space.
    for (int tile = wv; tile < 1024; tile += 16) {
        int ti = tile >> 5, tj = tile & 31;
        if (tj < ti) continue;               // only j > i can suppress
        int gi0 = ti * 16, gj0 = tj * 16;

        // WMMA outer-sum: D[m][n] = area_i[m] + area_j[n]  (scaled into f16)
        v16h a = {}; v16h bm = {};
        if (lane < 16) {
            a[0]  = (_Float16)(sarea[gi0 + lane] * AREA_SCL);  // A: K=0 col = area_i
            a[1]  = (_Float16)1.0f;                            // A: K=1 col = 1
            bm[0] = (_Float16)1.0f;                            // B: K=0 row = 1
            bm[1] = (_Float16)(sarea[gj0 + lane] * AREA_SCL);  // B: K=1 row = area_j
        }
        v8f acc = {};
        acc = __builtin_amdgcn_wmma_f32_16x16x32_f16(false, a, false, bm,
                                                     (short)0, acc, false, false);

        int nlo = lane & 15;
        float4 bj = sbox[gj0 + nlo];
        #pragma unroll
        for (int k = 0; k < 8; ++k) {
            int m  = k + ((lane >= 16) ? 8 : 0);    // C/D layout: VGPR k -> rows k, k+8
            int gi = gi0 + m, gj = gj0 + nlo;
            float4 bi = sbox[gi];
            float ix1 = fmaxf(bi.x, bj.x), iy1 = fmaxf(bi.y, bj.y);
            float ix2 = fminf(bi.z, bj.z), iy2 = fminf(bi.w, bj.w);
            float inter = fmaxf(ix2 - ix1, 0.f) * fmaxf(iy2 - iy1, 0.f);
            float uni = acc[k] * AREA_SCL_INV - inter;
            // division-free IoU threshold: inter > IOU_THR * max(uni, eps)
            bool p = (inter > 0.5f * fmaxf(uni, 1e-8f)) && (gj > gi);
            unsigned int bal = (unsigned int)__ballot(p);   // wave32 mask
            if (lane == 0) {
                unsigned int wd = (unsigned int)tj >> 1;
                unsigned int sh = (unsigned int)(tj & 1) * 16;
                atomicOr(&adj[gi0 + k    ][wd], (bal & 0xFFFFu) << sh);
                atomicOr(&adj[gi0 + k + 8][wd], (bal >> 16)     << sh);
            }
        }
    }
    __syncthreads();

    // Greedy suppression scan: wave 0, 16 lanes each own one 32-bit keep word.
    if (t < 32) {
        unsigned int keepw = 0xFFFFFFFFu;
        for (int i = 0; i < KSEL; ++i) {
            unsigned int kw = (unsigned int)__shfl((int)keepw, i >> 5, 32);
            bool alive = ((kw >> (i & 31)) & 1u) && (sval[i] > 0.5f);
            if (alive && lane < 16) keepw &= ~adj[i][lane];
            if (lane == 0) skept[i] = alive ? 1u : 0u;
        }
    }
    __syncthreads();

    if (t < KSEL)
        cand_score[(size_t)bc * KSEL + t] = skept[t] ? sval[t] : -1.f;
}

// ---------------------------------------------------------------------------
// K3: per-image top-1000 over 18*500 candidates. Full bitonic sort of 16384
// composite keys in LDS (128 KB, legal on 320 KB/WGP CDNA5 LDS).
// ---------------------------------------------------------------------------
__global__ __launch_bounds__(1024)
void final_topk_kernel(const float* __restrict__ cand_score,
                       const float* __restrict__ cand_box,
                       float* __restrict__ out_b,
                       float* __restrict__ out_s,
                       float* __restrict__ out_c,
                       float* __restrict__ out_nv) {
    int b = blockIdx.x;
    int t = threadIdx.x;
    __shared__ unsigned long long keys[SORTN];
    __shared__ int vcnt;
    if (t == 0) vcnt = 0;

    for (int i = t; i < SORTN; i += 1024) {
        unsigned long long kk = 0ull;
        if (i < FLAT_C) {
            float s = cand_score[(size_t)b * FLAT_C + i];
            kk = ((unsigned long long)fkey(s) << 32) |
                 (unsigned long long)(0xFFFFFFFFu - (unsigned int)i);  // idx asc on ties
        }
        keys[i] = kk;
    }
    __syncthreads();

    for (unsigned int k = 2; k <= SORTN; k <<= 1) {
        for (unsigned int j = k >> 1; j > 0; j >>= 1) {
            for (unsigned int idx = t; idx < SORTN; idx += 1024) {
                unsigned int ixj = idx ^ j;
                if (ixj > idx) {
                    bool desc = ((idx & k) == 0);
                    unsigned long long ka = keys[idx], kb = keys[ixj];
                    bool swp = desc ? (ka < kb) : (ka > kb);
                    if (swp) { keys[idx] = kb; keys[ixj] = ka; }
                }
            }
            __syncthreads();
        }
    }

    if (t < MAXDET) {
        unsigned long long e = keys[t];            // fully descending
        unsigned int fk = (unsigned int)(e >> 32);
        unsigned int i  = 0xFFFFFFFFu - (unsigned int)(e & 0xFFFFFFFFull);
        float s = fkey_inv(fk);
        bool valid = (s > 0.f) && (i < FLAT_C);
        float4 bx = valid ? ((const float4*)cand_box)[(size_t)b * FLAT_C + i]
                          : make_float4(0.f, 0.f, 0.f, 0.f);
        ((float4*)out_b)[b * MAXDET + t] = bx;
        out_s[b * MAXDET + t] = valid ? s : 0.f;
        out_c[b * MAXDET + t] = valid ? (float)(i / KSEL) : 0.f;
        if (valid) atomicAdd(&vcnt, 1);
    }
    __syncthreads();
    if (t == 0) out_nv[b] = (float)vcnt;
}

// ---------------------------------------------------------------------------
extern "C" void kernel_launch(void* const* d_in, const int* in_sizes, int n_in,
                              void* d_out, int out_size, void* d_ws, size_t ws_size,
                              hipStream_t stream) {
    const float* pred = (const float*)d_in[1];   // d_in[0] = images (unused)

    float* ws         = (float*)d_ws;
    float* boxes_dec  = ws;                                        // B*N*4
    float* scoresT    = boxes_dec + (size_t)BATCH * NANCH * 4;     // B*C*N
    float* cand_box   = scoresT   + (size_t)BATCH * NCLS * NANCH;  // B*C*500*4
    float* cand_score = cand_box  + (size_t)BATCH * NCLS * KSEL * 4; // B*C*500
    (void)ws_size; (void)in_sizes; (void)n_in; (void)out_size;

    float* out   = (float*)d_out;
    float* out_b = out;                             // [8,1000,4]
    float* out_s = out + BATCH * MAXDET * 4;        // [8,1000]
    float* out_c = out_s + BATCH * MAXDET;          // [8,1000]
    float* out_n = out_c + BATCH * MAXDET;          // [8]

    int total = BATCH * NANCH;
    decode_kernel<<<(total + 255) / 256, 256, 0, stream>>>(pred, boxes_dec, scoresT);
    topk_kernel<<<BATCH * NCLS, 1024, 0, stream>>>(scoresT, boxes_dec, cand_box, cand_score);
    nms_kernel<<<BATCH * NCLS, 512, 0, stream>>>(cand_box, cand_score);
    final_topk_kernel<<<BATCH, 1024, 0, stream>>>(cand_score, cand_box,
                                                  out_b, out_s, out_c, out_n);
}
